// VoxelBackBone8x_21680994910947
// MI455X (gfx1250) — matrix operand order repaired and checked by reading, not compiled, not run on hardware
//
#include <hip/hip_runtime.h>
#include <hip/hip_fp16.h>

#define EPSV 0.001f

typedef __attribute__((ext_vector_type(16))) _Float16 v16h;
typedef __attribute__((ext_vector_type(8)))  _Float16 h8;
typedef __attribute__((ext_vector_type(4)))  _Float16 h4;
typedef __attribute__((ext_vector_type(8)))  float    v8f;

// ---------------------------------------------------------------- utilities

__global__ void k_zero(float* __restrict__ p, long long n) {
  long long i  = (long long)blockIdx.x * blockDim.x + threadIdx.x;
  long long st = (long long)gridDim.x * blockDim.x;
  for (; i < n; i += st) p[i] = 0.f;
}

__global__ void k_scatter(const float* __restrict__ f, const int* __restrict__ c,
                          _Float16* __restrict__ x0, float* __restrict__ m1, int N) {
  int i = blockIdx.x * blockDim.x + threadIdx.x;
  if (i >= N) return;
  int z = c[i * 4 + 1], y = c[i * 4 + 2], x = c[i * 4 + 3];
  long long s = ((long long)z * 256 + y) * 256 + x;
  x0[s * 4 + 0] = (_Float16)f[i * 4 + 0];
  x0[s * 4 + 1] = (_Float16)f[i * 4 + 1];
  x0[s * 4 + 2] = (_Float16)f[i * 4 + 2];
  x0[s * 4 + 3] = (_Float16)f[i * 4 + 3];
  m1[s] = 1.f;
}

__global__ void k_sum(const float* __restrict__ p, float* __restrict__ out, long long n) {
  float l = 0.f;
  long long i  = (long long)blockIdx.x * blockDim.x + threadIdx.x;
  long long st = (long long)gridDim.x * blockDim.x;
  for (; i < n; i += st) l += p[i];
  __shared__ float red[256];
  red[threadIdx.x] = l;
  __syncthreads();
  for (int o = 128; o > 0; o >>= 1) {
    if ((int)threadIdx.x < o) red[threadIdx.x] += red[threadIdx.x + o];
    __syncthreads();
  }
  if (threadIdx.x == 0) atomicAdd(out, red[0]);
}

// Repack DHWIO f32 weights -> f16 WMMA B tiles with tap-packed K:
// logical K:  Kg = tap*CIN + cin, blocked 32-wide: tb = Kg/32.
// dst[(((jn*TB + tb)*16 + n)*32 + k32], co = jn*16+n.
__global__ void k_repack(const float* __restrict__ W, _Float16* __restrict__ dst,
                         int T, int CIN, int COUT) {
  int TB = (T * CIN + 31) / 32, NT = COUT >> 4;
  long long total = (long long)NT * TB * 512;
  long long i  = (long long)blockIdx.x * blockDim.x + threadIdx.x;
  long long st = (long long)gridDim.x * blockDim.x;
  for (; i < total; i += st) {
    int k32 = (int)(i & 31);
    long long r = i >> 5;
    int n = (int)(r & 15); r >>= 4;
    int tb = (int)(r % TB);
    int jn = (int)(r / TB);
    int Kg = tb * 32 + k32;
    int tap = Kg / CIN, cin = Kg % CIN;
    int co = jn * 16 + n;
    float v = (tap < T) ? W[((long long)tap * CIN + cin) * COUT + co] : 0.f;
    dst[i] = (_Float16)v;
  }
}

// mask' = (conv(mask, ones) > 0); fused active-site count into *ncnt.
__global__ void k_maskconv(const float* __restrict__ mi, float* __restrict__ mo,
                           float* __restrict__ ncnt,
                           int Di, int Hi, int Wi, int Do, int Ho, int Wo,
                           int kd, int kh, int kw, int sd, int sh, int sw,
                           int pd, int ph, int pw) {
  int Ms = Do * Ho * Wo;
  float local = 0.f;
  for (int s = blockIdx.x * blockDim.x + threadIdx.x; s < Ms;
       s += gridDim.x * blockDim.x) {
    int ow = s % Wo; int t = s / Wo; int oh = t % Ho; int od = t / Ho;
    float act = 0.f;
    for (int a = 0; a < kd; ++a) {
      int id = od * sd - pd + a; if (id < 0 || id >= Di) continue;
      for (int b = 0; b < kh; ++b) {
        int ih = oh * sh - ph + b; if (ih < 0 || ih >= Hi) continue;
        for (int cw = 0; cw < kw; ++cw) {
          int iw = ow * sw - pw + cw; if (iw < 0 || iw >= Wi) continue;
          if (mi[((long long)id * Hi + ih) * Wi + iw] > 0.f) act = 1.f;
        }
      }
    }
    mo[s] = act;
    local += act;
  }
  __shared__ float red[256];
  red[threadIdx.x] = local;
  __syncthreads();
  for (int o = 128; o > 0; o >>= 1) {
    if ((int)threadIdx.x < o) red[threadIdx.x] += red[threadIdx.x + o];
    __syncthreads();
  }
  if (threadIdx.x == 0) atomicAdd(ncnt, red[0]);
}

// ---------------------------------------------------------------- WMMA conv
// One wave32 computes 16 sites x (NTB*16) couts. K is tap-packed:
//   Kg = tap*CIN + cin, consumed in 32-wide blocks by v_wmma_f32_16x16x32_f16.
// Activations are stored f16 (NDHWC), so the A fragment is two aligned 16B
// loads + shuffle per k-block; no per-element converts.
// A layout (16-bit 16x32, ISA 7.12.2): per lane (row M = lane&15), half hi:
//   elements [0..7]  hold K32 = hi*8 + [0..7]
//   elements [8..15] hold K32 = 16 + hi*8 + [0..7]
template <int CIN, int KH, int KW, int NTB>
__global__ __launch_bounds__(256) void k_conv(
    const _Float16* __restrict__ src, const _Float16* __restrict__ wt,
    const float* __restrict__ maskO, float* __restrict__ dst,
    int Din, int Hin, int Win, int Dout, int Hout, int Wout, int Cout,
    int kd, int sd, int sh, int sw, int pd, int ph, int pw) {
  const int Ms = Dout * Hout * Wout;
  const int NTG = (Cout >> 4) / NTB;
  const int tilesM = (Ms + 15) >> 4;
  const long long wave = ((long long)blockIdx.x * blockDim.x + threadIdx.x) >> 5;
  if (wave >= (long long)tilesM * NTG) return;  // wave-uniform: EXEC all-1s
  const int lane = threadIdx.x & 31;
  const int jm = (int)(wave / NTG);
  const int jn = (int)(wave % NTG) * NTB;
  const int mrow = lane & 15;
  const int hi = lane >> 4;
  const int s = jm * 16 + mrow;
  const bool srow = s < Ms;
  int od = 0, oh = 0, ow = 0;
  if (srow) { ow = s % Wout; int t = s / Wout; oh = t % Hout; od = t / Hout; }
  const int taps = kd * KH * KW;
  const int TB = (taps * CIN + 31) / 32;

  v8f acc[NTB];
#pragma unroll
  for (int nb = 0; nb < NTB; ++nb) acc[nb] = (v8f){};

  // per-lane neighbor pointer for a given tap (src-base ptr when invalid)
  auto nbr = [&](int tap, bool& v) -> const _Float16* {
    int cw = tap % KW;
    int r = tap / KW;
    int bh = r % KH;
    int ad = r / KH;
    int id = od * sd - pd + ad;
    int ih = oh * sh - ph + bh;
    int iw = ow * sw - pw + cw;
    v = srow && (unsigned)id < (unsigned)Din && (unsigned)ih < (unsigned)Hin &&
        (unsigned)iw < (unsigned)Win && tap < taps;
    return src + (v ? (((size_t)id * Hin + ih) * Win + iw) * (size_t)CIN : 0);
  };
  auto ldh8 = [](const _Float16* p, bool v) -> h8 {
    h8 z = {};
    if (v) z = *(const h8*)p;
    return z;
  };
  auto ldh4 = [](const _Float16* p, bool v) -> h4 {
    h4 z = {};
    if (v) z = *(const h4*)p;
    return z;
  };

  const _Float16* wb = wt + (size_t)jn * TB * 512 + mrow * 32 + hi * 8;

  for (int tb = 0; tb < TB; ++tb) {
    v16h a;
    if constexpr (CIN >= 32) {
      // one tap per (CIN/32) blocks: tap = tb/KB, cin base = (tb%KB)*32
      constexpr int KB = CIN / 32;
      int tap = tb / KB;
      int cb = (tb % KB) * 32;
      bool v;
      const _Float16* sp = nbr(tap, v);
      h8 p0 = ldh8(sp + cb + hi * 8, v);
      h8 p1 = ldh8(sp + cb + 16 + hi * 8, v);
      a = __builtin_shufflevector(p0, p1, 0, 1, 2, 3, 4, 5, 6, 7,
                                  8, 9, 10, 11, 12, 13, 14, 15);
    } else if constexpr (CIN == 16) {
      // two taps per block: K32<16 -> tap 2tb, K32>=16 -> tap 2tb+1
      bool v0, v1;
      const _Float16* sp0 = nbr(2 * tb + 0, v0);
      const _Float16* sp1 = nbr(2 * tb + 1, v1);
      h8 p0 = ldh8(sp0 + hi * 8, v0);
      h8 p1 = ldh8(sp1 + hi * 8, v1);
      a = __builtin_shufflevector(p0, p1, 0, 1, 2, 3, 4, 5, 6, 7,
                                  8, 9, 10, 11, 12, 13, 14, 15);
    } else {
      // CIN == 4: eight taps per block; this lane-half holds taps
      // tb*8 + {hi*2, hi*2+1, 4+hi*2, 5+hi*2}, 4 channels each.
      bool v0, v1, v2, v3;
      const _Float16* sp0 = nbr(tb * 8 + hi * 2 + 0, v0);
      const _Float16* sp1 = nbr(tb * 8 + hi * 2 + 1, v1);
      const _Float16* sp2 = nbr(tb * 8 + hi * 2 + 4, v2);
      const _Float16* sp3 = nbr(tb * 8 + hi * 2 + 5, v3);
      h4 q0 = ldh4(sp0, v0), q1 = ldh4(sp1, v1);
      h4 q2 = ldh4(sp2, v2), q3 = ldh4(sp3, v3);
      h8 lo = __builtin_shufflevector(q0, q1, 0, 1, 2, 3, 4, 5, 6, 7);
      h8 hv = __builtin_shufflevector(q2, q3, 0, 1, 2, 3, 4, 5, 6, 7);
      a = __builtin_shufflevector(lo, hv, 0, 1, 2, 3, 4, 5, 6, 7,
                                  8, 9, 10, 11, 12, 13, 14, 15);
    }
    const _Float16* wtap = wb + (size_t)tb * 512;
#pragma unroll
    for (int nb = 0; nb < NTB; ++nb) {
      const _Float16* bp = wtap + (size_t)nb * TB * 512;
      h8 b0 = *(const h8*)bp;
      h8 b1 = *(const h8*)(bp + 16);
      v16h b = __builtin_shufflevector(b0, b1, 0, 1, 2, 3, 4, 5, 6, 7,
                                       8, 9, 10, 11, 12, 13, 14, 15);
      acc[nb] = __builtin_amdgcn_wmma_f32_16x16x32_f16(
          false, a, false, b, (short)0, acc[nb], false, false);
    }
  }

  // C/D layout: lane<16 -> rows 0..7, lane>=16 -> rows 8..15; col N = lane&15.
#pragma unroll
  for (int r = 0; r < 8; ++r) {
    int row = hi * 8 + r;
    int so = jm * 16 + row;
    if (so < Ms) {
      float mval = maskO[so];
#pragma unroll
      for (int nb = 0; nb < NTB; ++nb)
        dst[(size_t)so * Cout + (jn + nb) * 16 + mrow] = acc[nb][r] * mval;
    }
  }
}

// ---------------------------------------------------------------- BN pieces

__global__ __launch_bounds__(256) void k_stats(const float* __restrict__ y,
                                               float* __restrict__ sums,
                                               int Ms, int C) {
  const int t = threadIdx.x;
  const int c = t % C;
  const int slice = t / C;
  const int G = 256 / C;
  float s = 0.f, q = 0.f;
  for (long long i = (long long)blockIdx.x * G + slice; i < Ms;
       i += (long long)gridDim.x * G) {
    float v = y[i * (long long)C + c];
    s += v;
    q += v * v;
  }
  __shared__ float ls[256], lq[256];
  ls[t] = s; lq[t] = q;
  __syncthreads();
  for (int off = 128; off >= C; off >>= 1) {
    if (t < off) { ls[t] += ls[t + off]; lq[t] += lq[t + off]; }
    __syncthreads();
  }
  if (t < C) {
    atomicAdd(&sums[c], ls[t]);
    atomicAdd(&sums[C + c], lq[t]);
  }
}

__global__ void k_finalize(const float* __restrict__ sums,
                           const float* __restrict__ nbuf,
                           const float* __restrict__ g,
                           const float* __restrict__ b,
                           float* __restrict__ sc, int C) {
  int c = threadIdx.x;
  if (c >= C) return;
  float n = fmaxf(nbuf[0], 1.f);
  float mean = sums[c] / n;
  float var = fmaxf(sums[C + c] / n - mean * mean, 0.f);
  float scale = g[c] * rsqrtf(var + EPSV);
  sc[c] = scale;
  sc[C + c] = b[c] - mean * scale;
}

// BN + ReLU + mask; emits next layer's f16 activation.
__global__ void k_apply_h(const float* __restrict__ y, const float* __restrict__ mask,
                          const float* __restrict__ sc, _Float16* __restrict__ out,
                          long long Ms, int C) {
  long long total = Ms * C;
  long long i  = (long long)blockIdx.x * blockDim.x + threadIdx.x;
  long long st = (long long)gridDim.x * blockDim.x;
  for (; i < total; i += st) {
    long long s = i / C;
    int c = (int)(i % C);
    float v = y[i] * sc[c] + sc[C + c];
    out[i] = (_Float16)(fmaxf(v, 0.f) * mask[s]);
  }
}

// Final layer: BN + ReLU + mask in place on f32 output.
__global__ void k_apply(float* __restrict__ y, const float* __restrict__ mask,
                        const float* __restrict__ sc, long long Ms, int C) {
  long long total = Ms * C;
  long long i  = (long long)blockIdx.x * blockDim.x + threadIdx.x;
  long long st = (long long)gridDim.x * blockDim.x;
  for (; i < total; i += st) {
    long long s = i / C;
    int c = (int)(i % C);
    float v = y[i] * sc[c] + sc[C + c];
    y[i] = fmaxf(v, 0.f) * mask[s];
  }
}

// ---------------------------------------------------------------- host side

struct LD {
  int Cin, Cout;
  int Di, Hi, Wi, Do, Ho, Wo;
  int kd, kh, kw, sd, sh, sw, pd, ph, pw;
  int subm;
};

static inline int ngrid(long long work, int per, int cap) {
  long long b = (work + per - 1) / per;
  if (b < 1) b = 1;
  if (b > cap) b = cap;
  return (int)b;
}

extern "C" void kernel_launch(void* const* d_in, const int* in_sizes, int n_in,
                              void* d_out, int out_size, void* d_ws,
                              size_t ws_size, hipStream_t stream) {
  (void)in_sizes; (void)n_in; (void)out_size;
  const float* feats  = (const float*)d_in[0];
  const int*   coords = (const int*)d_in[1];
  const int NVOX = 50000;

  static const LD L[12] = {
      {  4,  16, 33, 256, 256, 33, 256, 256, 3, 3, 3, 1, 1, 1, 1, 1, 1, 1},
      { 16,  16, 33, 256, 256, 33, 256, 256, 3, 3, 3, 1, 1, 1, 1, 1, 1, 1},
      { 16,  32, 33, 256, 256, 17, 128, 128, 3, 3, 3, 2, 2, 2, 1, 1, 1, 0},
      { 32,  32, 17, 128, 128, 17, 128, 128, 3, 3, 3, 1, 1, 1, 1, 1, 1, 1},
      { 32,  32, 17, 128, 128, 17, 128, 128, 3, 3, 3, 1, 1, 1, 1, 1, 1, 1},
      { 32,  64, 17, 128, 128,  9,  64,  64, 3, 3, 3, 2, 2, 2, 1, 1, 1, 0},
      { 64,  64,  9,  64,  64,  9,  64,  64, 3, 3, 3, 1, 1, 1, 1, 1, 1, 1},
      { 64,  64,  9,  64,  64,  9,  64,  64, 3, 3, 3, 1, 1, 1, 1, 1, 1, 1},
      { 64,  64,  9,  64,  64,  4,  32,  32, 3, 3, 3, 2, 2, 2, 0, 1, 1, 0},
      { 64,  64,  4,  32,  32,  4,  32,  32, 3, 3, 3, 1, 1, 1, 1, 1, 1, 1},
      { 64,  64,  4,  32,  32,  4,  32,  32, 3, 3, 3, 1, 1, 1, 1, 1, 1, 1},
      { 64, 128,  4,  32,  32,  1,  32,  32, 3, 1, 1, 2, 1, 1, 0, 0, 0, 0},
  };
  static const int mslot[12] = {0, 0, 1, 1, 1, 2, 2, 2, 3, 3, 3, 4};
  static const int ntbs[12]  = {1, 1, 2, 2, 2, 4, 4, 4, 4, 4, 4, 4};

  const size_t S1 = 33ULL * 256 * 256, S2 = 17ULL * 128 * 128,
               S3 = 9ULL * 64 * 64, S4 = 4ULL * 32 * 32, S5 = 1024;

  float* ws = (float*)d_ws;
  size_t cur = 0;
  auto A = [&](size_t n) { size_t o = cur; cur += (n + 15) & ~(size_t)15; return o; };
  // f32 regions
  size_t oY  = A(S1 * 16);                      // shared conv-output scratch
  size_t oM1 = A(S1), oM2 = A(S2), oM3 = A(S3), oM4 = A(S4), oM5 = A(S5);
  size_t oST = A(256), oSC = A(256), oNB = A(16);
  size_t oWP = A(380000);                       // f16 weight-pack region
  // f16 activation regions (sizes given in f16 elements, stored as /2 floats)
  size_t oX0h = A(S1 * 4 / 2);
  size_t oH1a = A(S1 * 16 / 2), oH1b = A(S1 * 16 / 2);
  size_t oH2a = A(S2 * 32 / 2), oH2b = A(S2 * 32 / 2);
  size_t oH3a = A(S3 * 64 / 2), oH3b = A(S3 * 64 / 2);
  size_t oH4a = A(S4 * 64 / 2), oH4b = A(S4 * 64 / 2);
  if (ws_size < cur * sizeof(float)) return;  // workspace too small; bail

  float* Y  = ws + oY;
  float* M1 = ws + oM1;
  float* ST = ws + oST; float* SC = ws + oSC; float* NB = ws + oNB;
  _Float16* WP = (_Float16*)(ws + oWP);
  _Float16* X0h = (_Float16*)(ws + oX0h);

  // f16 activation ping-pong: hact[li] is layer li's input
  _Float16* hact[12];
  hact[0] = X0h;
  hact[1] = (_Float16*)(ws + oH1a);  hact[2] = (_Float16*)(ws + oH1b);
  hact[3] = (_Float16*)(ws + oH2a);  hact[4] = (_Float16*)(ws + oH2b);
  hact[5] = (_Float16*)(ws + oH2a);
  hact[6] = (_Float16*)(ws + oH3a);  hact[7] = (_Float16*)(ws + oH3b);
  hact[8] = (_Float16*)(ws + oH3a);
  hact[9] = (_Float16*)(ws + oH4a);  hact[10] = (_Float16*)(ws + oH4b);
  hact[11] = (_Float16*)(ws + oH4a);
  float* mbuf[5] = {M1, ws + oM2, ws + oM3, ws + oM4, ws + oM5};

  // ---- build dense f16 input + mask
  k_zero<<<ngrid((long long)S1 * 2, 256, 8192), 256, 0, stream>>>(
      (float*)X0h, (long long)S1 * 2);
  k_zero<<<ngrid((long long)S1, 256, 8192), 256, 0, stream>>>(M1, (long long)S1);
  k_zero<<<1, 32, 0, stream>>>(NB, 16);
  k_scatter<<<(NVOX + 255) / 256, 256, 0, stream>>>(feats, coords, X0h, M1, NVOX);
  k_sum<<<ngrid((long long)S1, 256 * 8, 2048), 256, 0, stream>>>(M1, NB + 0, (long long)S1);

  // ---- repack all weights to f16 WMMA-B layout (tap-packed K)
  _Float16* wp[12];
  size_t wcur = 0;
  for (int li = 0; li < 12; ++li) {
    int T = L[li].kd * L[li].kh * L[li].kw;
    int TB = (T * L[li].Cin + 31) / 32, NT = L[li].Cout / 16;
    size_t n = (size_t)NT * TB * 512;
    wp[li] = WP + wcur;
    k_repack<<<ngrid((long long)n, 256, 512), 256, 0, stream>>>(
        (const float*)d_in[2 + 3 * li], wp[li], T, L[li].Cin, L[li].Cout);
    wcur += n;
  }

  // ---- run the 12-layer pipeline
  for (int li = 0; li < 12; ++li) {
    const LD& P = L[li];
    const int so = mslot[li];
    float* mO = mbuf[so];
    const int Ms = P.Do * P.Ho * P.Wo;

    if (!P.subm) {
      k_maskconv<<<ngrid(Ms, 256, 8192), 256, 0, stream>>>(
          mbuf[so - 1], mO, NB + so, P.Di, P.Hi, P.Wi, P.Do, P.Ho, P.Wo,
          P.kd, P.kh, P.kw, P.sd, P.sh, P.sw, P.pd, P.ph, P.pw);
    }

    long long waves = (long long)((Ms + 15) / 16) * (P.Cout / 16 / ntbs[li]);
    int cblocks = (int)((waves * 32 + 255) / 256);
    const _Float16* src = hact[li];
    float* dst = (li == 11) ? (float*)d_out : Y;

#define CONV_ARGS src, wp[li], mO, dst, P.Di, P.Hi, P.Wi, P.Do, P.Ho, P.Wo, \
                  P.Cout, P.kd, P.sd, P.sh, P.sw, P.pd, P.ph, P.pw
    switch (li) {
      case 0:  k_conv<4, 3, 3, 1><<<cblocks, 256, 0, stream>>>(CONV_ARGS); break;
      case 1:  k_conv<16, 3, 3, 1><<<cblocks, 256, 0, stream>>>(CONV_ARGS); break;
      case 2:  k_conv<16, 3, 3, 2><<<cblocks, 256, 0, stream>>>(CONV_ARGS); break;
      case 3:
      case 4:  k_conv<32, 3, 3, 2><<<cblocks, 256, 0, stream>>>(CONV_ARGS); break;
      case 5:  k_conv<32, 3, 3, 4><<<cblocks, 256, 0, stream>>>(CONV_ARGS); break;
      case 11: k_conv<64, 1, 1, 4><<<cblocks, 256, 0, stream>>>(CONV_ARGS); break;
      default: k_conv<64, 3, 3, 4><<<cblocks, 256, 0, stream>>>(CONV_ARGS); break;
    }
#undef CONV_ARGS

    // masked batch-norm + ReLU
    k_zero<<<1, 256, 0, stream>>>(ST, 256);
    long long tot = (long long)Ms * P.Cout;
    k_stats<<<ngrid(tot, 256 * 8, 2048), 256, 0, stream>>>(dst, ST, Ms, P.Cout);
    k_finalize<<<1, 128, 0, stream>>>(ST, NB + so,
                                      (const float*)d_in[2 + 3 * li + 1],
                                      (const float*)d_in[2 + 3 * li + 2],
                                      SC, P.Cout);
    if (li == 11) {
      k_apply<<<ngrid(tot, 256, 16384), 256, 0, stream>>>(dst, mO, SC,
                                                          (long long)Ms, P.Cout);
    } else {
      k_apply_h<<<ngrid(tot, 256, 16384), 256, 0, stream>>>(
          dst, mO, SC, hact[li + 1], (long long)Ms, P.Cout);
    }
  }
}
